// SchNetModel_81844896792896
// MI455X (gfx1250) — matrix operand order repaired and checked by reading, not compiled, not run on hardware
//
#include <hip/hip_runtime.h>
#include <hip/hip_bf16.h>
#include <math.h>

typedef __attribute__((ext_vector_type(16))) _Float16 v16h;
typedef __attribute__((ext_vector_type(8)))  _Float16 v8h;
typedef __attribute__((ext_vector_type(8)))  float    v8f;

#define N_NODES  50000
#define N_EDGES  1000000
#define N_GRAPHS 2500
#define DIM      64
#define FAN      50
#define GAPF     (5.0f / 49.0f)
#define INV_GAP  (49.0f / 5.0f)
#define LOG2F_C  0.6931471805599453f

// ---------------------------------------------------------------------------
// Activations (match reference softplus_b05 / shift_softplus)
// ---------------------------------------------------------------------------
__device__ __forceinline__ float softplus_b05(float x) {
    float hx = 0.5f * x;
    return (hx > 14.0f) ? x : 2.0f * log1pf(__expf(hx));
}
__device__ __forceinline__ float shift_softplus(float x) {
    return (x > 15.0f) ? (x - LOG2F_C) : (log1pf(__expf(x)) - LOG2F_C);
}

// ---------------------------------------------------------------------------
// WMMA 16x16x32 f16->f32 (wave32)
// A 16x32: lanes 0-15 row=lane, K {kb+0..7, kb+16..23};
//          lanes 16-31 row=lane-16, K {kb+8..15, kb+24..31}
// B 32x16: lane%16 = col, same K split as A (lanes 0-15 K {kb+0..7, kb+16..23}, lanes 16-31 K {kb+8..15, kb+24..31})
// C/D f32: lane%16 = col, 8 VGPRs = rows (lane<16: M 0..7, else M 8..15)
// ---------------------------------------------------------------------------
__device__ __forceinline__ v8f wmma_f16(v16h a, v16h b, v8f c) {
    return __builtin_amdgcn_wmma_f32_16x16x32_f16(false, a, false, b,
                                                  (short)0, c, false, false);
}

// A fragment from a 16x64 row-major f16 LDS tile: two contiguous 16B reads
__device__ __forceinline__ v16h load_A_frag(const _Float16* T, int kbase) {
    int lane  = threadIdx.x & 31;
    int row   = lane & 15;
    int khalf = (lane >> 4) & 1;
    const _Float16* rp = T + row * 64 + kbase + khalf * 8;
    v8h x = *(const v8h*)rp;
    v8h y = *(const v8h*)(rp + 16);
    return __builtin_shufflevector(x, y, 0, 1, 2, 3, 4, 5, 6, 7,
                                   8, 9, 10, 11, 12, 13, 14, 15);
}

// ---------------------------------------------------------------------------
// Packed B-fragment weight storage in LDS: [8 tiles][32 lanes][16 halfs]
// tile t = kt*4 + nt  (kt in 0..1 -> K base kt*32, nt in 0..3 -> N base nt*16)
// Each lane's fragment is one contiguous 32-byte chunk.
// ---------------------------------------------------------------------------
__device__ __forceinline__ void pack_B(_Float16* __restrict__ P,
                                       const float* __restrict__ W,
                                       int tid, int krows) {
    for (int i = tid; i < 4096; i += 256) {
        int t = i >> 9;
        int l = (i >> 4) & 31;
        int e = i & 15;
        int K = ((t >> 2) << 5) + ((l >> 4) << 3) + (e & 7) + ((e >> 3) << 4);
        int N = ((t & 3) << 4) + (l & 15);
        P[i] = (_Float16)((K < krows) ? W[K * 64 + N] : 0.0f);
    }
}
__device__ __forceinline__ v16h load_B_pk(const _Float16* P, int kt, int nt) {
    int lane = threadIdx.x & 31;
    return *(const v16h*)(P + (((((kt << 2) | nt) << 5) + lane) << 4));
}

// RBF A fragment computed directly in registers from this lane's row distance
__device__ __forceinline__ v16h rbf_A_frag(float d, int kbase) {
    int lane  = threadIdx.x & 31;
    int khalf = (lane >> 4) & 1;
    v16h a;
#pragma unroll
    for (int e = 0; e < 16; ++e) {
        int k = kbase + khalf * 8 + (e & 7) + ((e >> 3) << 4);
        float r = 0.0f;
        if (k < FAN) {
            float t = d - (float)k * GAPF;
            r = __expf(-INV_GAP * t * t);
        }
        a[e] = (_Float16)r;
    }
    return a;
}

// ---------------------------------------------------------------------------
__global__ void zero_kernel(float* __restrict__ p, int n) {
    int i = blockIdx.x * 256 + threadIdx.x;
    if (i < n) p[i] = 0.0f;
}

__global__ void embed_kernel(const int* __restrict__ type, const float* __restrict__ emb,
                             float* __restrict__ node) {
    int i = blockIdx.x * 256 + threadIdx.x;
    if (i < N_NODES * DIM) {
        int r = i >> 6;
        node[i] = emb[type[r] * DIM + (i & 63)];
    }
}

// ---------------------------------------------------------------------------
// new_node = node @ W1  (no bias)
// ---------------------------------------------------------------------------
__global__ void node_gemm_kernel(const float* __restrict__ X, const float* __restrict__ W,
                                 float* __restrict__ Y, int nrows) {
    __shared__ __align__(32) _Float16 Wl[4096];
    __shared__ __align__(16) _Float16 st[8][16 * 64];
    int tid = threadIdx.x, lane = tid & 31, w = tid >> 5;
    pack_B(Wl, W, tid, 64);
    int rb = blockIdx.x * 128 + w * 16;
    bool full = (rb + 16 <= nrows);
    for (int idx = lane; idx < 16 * 64; idx += 32) {
        int r = rb + (idx >> 6);
        float v = full ? X[r * 64 + (idx & 63)]
                       : ((r < nrows) ? X[r * 64 + (idx & 63)] : 0.0f);
        st[w][idx] = (_Float16)v;
    }
    __syncthreads();
    v16h a0 = load_A_frag(st[w], 0);
    v16h a1 = load_A_frag(st[w], 32);
    int col0  = lane & 15;
    int rbase = (lane >> 4) << 3;
#pragma unroll
    for (int nt = 0; nt < 4; ++nt) {
        v8f c = {};
        c = wmma_f16(a0, load_B_pk(Wl, 0, nt), c);
        c = wmma_f16(a1, load_B_pk(Wl, 1, nt), c);
        int col = col0 + nt * 16;
        if (full) {
#pragma unroll
            for (int v = 0; v < 8; ++v)
                Y[(rb + rbase + v) * 64 + col] = c[v];
        } else {
#pragma unroll
            for (int v = 0; v < 8; ++v) {
                int row = rb + rbase + v;
                if (row < nrows) Y[row * 64 + col] = c[v];
            }
        }
    }
}

// ---------------------------------------------------------------------------
// Fused edge conv: filt = sp05(rbf(d)@Wf1+bf1)@Wf2+bf2 ; cf[dst] += nnew[src]*filt
// 128 edges per block, 16 per wave.
// ---------------------------------------------------------------------------
__global__ void edge_kernel(const float* __restrict__ dist,
                            const int* __restrict__ esrc, const int* __restrict__ edst,
                            const float* __restrict__ nnew,
                            const float* __restrict__ Wf1, const float* __restrict__ bf1,
                            const float* __restrict__ Wf2, const float* __restrict__ bf2,
                            float* __restrict__ cf) {
    __shared__ __align__(32) _Float16 F1[4096];   // Wf1 packed, K 50 -> 64 zero-pad
    __shared__ __align__(32) _Float16 F2[4096];
    __shared__ float b1l[64], b2l[64];
    __shared__ __align__(16) _Float16 st[8][16 * 64];
    __shared__ __align__(16) float dsh[128];

    int tid = threadIdx.x, lane = tid & 31, w = tid >> 5;
    int eb0 = blockIdx.x * 128;

    // Async-stage this block's 128 distances into LDS (gfx1250 async path),
    // and prefetch the edge index arrays.
    if (tid < 128) {
        int e = eb0 + tid;
        int ec = (e < N_EDGES) ? e : (N_EDGES - 1);
        __builtin_prefetch(&esrc[ec], 0, 0);
        __builtin_prefetch(&edst[ec], 0, 0);
        const float* gp = dist + ec;
        unsigned lds_addr = (unsigned)(unsigned long long)(const void*)&dsh[tid];
        asm volatile("global_load_async_to_lds_b32 %0, %1, off"
                     :: "v"(lds_addr), "v"(gp) : "memory");
    }
    pack_B(F1, Wf1, tid, FAN);
    pack_B(F2, Wf2, tid, 64);
    if (tid < 64) { b1l[tid] = bf1[tid]; b2l[tid] = bf2[tid]; }
    asm volatile("s_wait_asynccnt 0x0" ::: "memory");
    __syncthreads();

    int eb    = eb0 + w * 16;
    int myrow = lane & 15;
    int rbase = (lane >> 4) << 3;
    bool full = (eb + 16 <= N_EDGES);
    float d = dsh[w * 16 + myrow];

    v16h a0 = rbf_A_frag(d, 0);
    v16h a1 = rbf_A_frag(d, 32);

    // h = softplus05(rbf @ Wf1 + bf1) -> staged f16
#pragma unroll
    for (int nt = 0; nt < 4; ++nt) {
        v8f c = {};
        c = wmma_f16(a0, load_B_pk(F1, 0, nt), c);
        c = wmma_f16(a1, load_B_pk(F1, 1, nt), c);
        int col = (lane & 15) + nt * 16;
        float bb = b1l[col];
#pragma unroll
        for (int v = 0; v < 8; ++v)
            st[w][(rbase + v) * 64 + col] = (_Float16)softplus_b05(c[v] + bb);
    }
    __syncthreads();
    v16h h0 = load_A_frag(st[w], 0);
    v16h h1 = load_A_frag(st[w], 32);

    // filt tiles; fused gather-multiply-scatter straight from C layout
#pragma unroll
    for (int nt = 0; nt < 4; ++nt) {
        v8f c = {};
        c = wmma_f16(h0, load_B_pk(F2, 0, nt), c);
        c = wmma_f16(h1, load_B_pk(F2, 1, nt), c);
        int col = (lane & 15) + nt * 16;
        float bb = b2l[col];
        if (full) {
#pragma unroll
            for (int v = 0; v < 8; ++v) {
                int e  = eb + rbase + v;
                int s  = esrc[e];
                int dd = edst[e];
                float x = nnew[s * 64 + col];
                atomicAdd(&cf[dd * 64 + col], (c[v] + bb) * x);
            }
        } else {
#pragma unroll
            for (int v = 0; v < 8; ++v) {
                int e = eb + rbase + v;
                if (e < N_EDGES) {
                    int s  = esrc[e];
                    int dd = edst[e];
                    float x = nnew[s * 64 + col];
                    atomicAdd(&cf[dd * 64 + col], (c[v] + bb) * x);
                }
            }
        }
    }
}

// ---------------------------------------------------------------------------
// node += softplus05(cf @ W2 + b2) @ W3 + b3
// ---------------------------------------------------------------------------
__global__ void node_update_kernel(const float* __restrict__ cf,
                                   const float* __restrict__ W2, const float* __restrict__ b2,
                                   const float* __restrict__ W3, const float* __restrict__ b3,
                                   float* __restrict__ node, int nrows) {
    __shared__ __align__(32) _Float16 W2l[4096], W3l[4096];
    __shared__ float b2l[64], b3l[64];
    __shared__ __align__(16) _Float16 st[8][16 * 64];
    int tid = threadIdx.x, lane = tid & 31, w = tid >> 5;
    pack_B(W2l, W2, tid, 64);
    pack_B(W3l, W3, tid, 64);
    if (tid < 64) { b2l[tid] = b2[tid]; b3l[tid] = b3[tid]; }
    int rb = blockIdx.x * 128 + w * 16;
    bool full = (rb + 16 <= nrows);
    for (int idx = lane; idx < 16 * 64; idx += 32) {
        int r = rb + (idx >> 6);
        float v = full ? cf[r * 64 + (idx & 63)]
                       : ((r < nrows) ? cf[r * 64 + (idx & 63)] : 0.0f);
        st[w][idx] = (_Float16)v;
    }
    __syncthreads();
    v16h a0 = load_A_frag(st[w], 0);
    v16h a1 = load_A_frag(st[w], 32);
    int rbase = (lane >> 4) << 3;
#pragma unroll
    for (int nt = 0; nt < 4; ++nt) {
        v8f c = {};
        c = wmma_f16(a0, load_B_pk(W2l, 0, nt), c);
        c = wmma_f16(a1, load_B_pk(W2l, 1, nt), c);
        int col = (lane & 15) + nt * 16;
        float bb = b2l[col];
#pragma unroll
        for (int v = 0; v < 8; ++v)
            st[w][(rbase + v) * 64 + col] = (_Float16)softplus_b05(c[v] + bb);
    }
    __syncthreads();
    v16h h0 = load_A_frag(st[w], 0);
    v16h h1 = load_A_frag(st[w], 32);
#pragma unroll
    for (int nt = 0; nt < 4; ++nt) {
        v8f c = {};
        c = wmma_f16(h0, load_B_pk(W3l, 0, nt), c);
        c = wmma_f16(h1, load_B_pk(W3l, 1, nt), c);
        int col = (lane & 15) + nt * 16;
        float bb = b3l[col];
        if (full) {
#pragma unroll
            for (int v = 0; v < 8; ++v)
                node[(rb + rbase + v) * 64 + col] += c[v] + bb;
        } else {
#pragma unroll
            for (int v = 0; v < 8; ++v) {
                int row = rb + rbase + v;
                if (row < nrows) node[row * 64 + col] += c[v] + bb;
            }
        }
    }
}

// ---------------------------------------------------------------------------
// out[g] += sum_rows( shift_softplus(node @ Wd1 + bd1) @ Wd2 + bd2 )
// ---------------------------------------------------------------------------
__global__ void readout_kernel(const float* __restrict__ node,
                               const float* __restrict__ Wd1, const float* __restrict__ bd1,
                               const float* __restrict__ Wd2, const float* __restrict__ bd2,
                               const int* __restrict__ gid, float* __restrict__ out,
                               int nrows) {
    __shared__ __align__(32) _Float16 Wl[4096];
    __shared__ float b1l[64], w2l[64];
    __shared__ __align__(16) _Float16 st[8][16 * 64];
    int tid = threadIdx.x, lane = tid & 31, w = tid >> 5;
    pack_B(Wl, Wd1, tid, 64);
    if (tid < 64) { b1l[tid] = bd1[tid]; w2l[tid] = Wd2[tid]; }
    int rb = blockIdx.x * 128 + w * 16;
    bool full = (rb + 16 <= nrows);
    for (int idx = lane; idx < 16 * 64; idx += 32) {
        int r = rb + (idx >> 6);
        float v = full ? node[r * 64 + (idx & 63)]
                       : ((r < nrows) ? node[r * 64 + (idx & 63)] : 0.0f);
        st[w][idx] = (_Float16)v;
    }
    __syncthreads();
    v16h a0 = load_A_frag(st[w], 0);
    v16h a1 = load_A_frag(st[w], 32);
    int rbase = (lane >> 4) << 3;
    float p[8];
#pragma unroll
    for (int v = 0; v < 8; ++v) p[v] = 0.0f;
#pragma unroll
    for (int nt = 0; nt < 4; ++nt) {
        v8f c = {};
        c = wmma_f16(a0, load_B_pk(Wl, 0, nt), c);
        c = wmma_f16(a1, load_B_pk(Wl, 1, nt), c);
        int col = (lane & 15) + nt * 16;
        float bb = b1l[col], wv = w2l[col];
#pragma unroll
        for (int v = 0; v < 8; ++v)
            p[v] += shift_softplus(c[v] + bb) * wv;
    }
#pragma unroll
    for (int m = 1; m < 16; m <<= 1) {
#pragma unroll
        for (int v = 0; v < 8; ++v) p[v] += __shfl_xor(p[v], m, 32);
    }
    float bd2v = bd2[0];
    if ((lane & 15) == 0) {       // lane 0 -> rows 0..7, lane 16 -> rows 8..15
#pragma unroll
        for (int v = 0; v < 8; ++v) {
            int row = rb + rbase + v;
            if (row < nrows) atomicAdd(&out[gid[row]], p[v] + bd2v);
        }
    }
}

// ---------------------------------------------------------------------------
extern "C" void kernel_launch(void* const* d_in, const int* in_sizes, int n_in,
                              void* d_out, int out_size, void* d_ws, size_t ws_size,
                              hipStream_t stream) {
    const int*   node_type = (const int*)d_in[0];
    const int*   src       = (const int*)d_in[1];
    const int*   dst       = (const int*)d_in[2];
    const int*   graph_ids = (const int*)d_in[3];
    const float* distance  = (const float*)d_in[4];
    const float* emb       = (const float*)d_in[5];
    const float* W1        = (const float*)d_in[6];
    const float* Wf1       = (const float*)d_in[7];
    const float* bf1       = (const float*)d_in[8];
    const float* Wf2       = (const float*)d_in[9];
    const float* bf2       = (const float*)d_in[10];
    const float* W2        = (const float*)d_in[11];
    const float* b2        = (const float*)d_in[12];
    const float* W3        = (const float*)d_in[13];
    const float* b3        = (const float*)d_in[14];
    const float* Wd1       = (const float*)d_in[15];
    const float* bd1       = (const float*)d_in[16];
    const float* Wd2       = (const float*)d_in[17];
    const float* bd2       = (const float*)d_in[18];
    float* out = (float*)d_out;

    float* node = (float*)d_ws;                 // [N_NODES * 64]
    float* nnew = node + (size_t)N_NODES * 64;  // [N_NODES * 64]
    float* cf   = nnew + (size_t)N_NODES * 64;  // [N_NODES * 64]

    const int nelem     = N_NODES * DIM;
    const int grid_el   = (nelem + 255) / 256;
    const int grid_node = (N_NODES + 127) / 128;
    const int grid_edge = (N_EDGES + 127) / 128;

    embed_kernel<<<grid_el, 256, 0, stream>>>(node_type, emb, node);

    for (int i = 0; i < 3; ++i) {
        zero_kernel<<<grid_el, 256, 0, stream>>>(cf, nelem);
        node_gemm_kernel<<<grid_node, 256, 0, stream>>>(node, W1 + (size_t)i * 64 * 64,
                                                        nnew, N_NODES);
        edge_kernel<<<grid_edge, 256, 0, stream>>>(distance, src, dst, nnew,
                                                   Wf1 + (size_t)i * FAN * 64,
                                                   bf1 + (size_t)i * 64,
                                                   Wf2 + (size_t)i * 64 * 64,
                                                   bf2 + (size_t)i * 64, cf);
        node_update_kernel<<<grid_node, 256, 0, stream>>>(cf,
                                                          W2 + (size_t)i * 64 * 64,
                                                          b2 + (size_t)i * 64,
                                                          W3 + (size_t)i * 64 * 64,
                                                          b3 + (size_t)i * 64,
                                                          node, N_NODES);
    }

    zero_kernel<<<(N_GRAPHS + 255) / 256, 256, 0, stream>>>(out, N_GRAPHS);
    readout_kernel<<<grid_node, 256, 0, stream>>>(node, Wd1, bd1, Wd2, bd2,
                                                  graph_ids, out, N_NODES);
}
